// ImportanceGuidedCrossAttention_47364899340798
// MI455X (gfx1250) — matrix operand order repaired and checked by reading, not compiled, not run on hardware
//
#include <hip/hip_runtime.h>

typedef __attribute__((ext_vector_type(16))) __bf16 v16bf;
typedef __attribute__((ext_vector_type(8)))  __bf16 v8bf;
typedef __attribute__((ext_vector_type(8)))  float  v8f;

union AFrag { v16bf v; v8bf h[2]; };

static constexpr int   BATCH  = 2;
static constexpr int   NQ     = 1024;
static constexpr int   MKV    = 4096;
static constexpr int   QDIM   = 1024;
static constexpr int   CDIM   = 1024;
static constexpr int   HEADS_ = 16;
static constexpr int   DHEAD  = 64;
static constexpr int   INNER_ = 1024;
static constexpr float SCALE_ = 0.125f;             // 64^-0.5
static constexpr float LOG2E  = 1.4426950408889634f;

// ---------------- conversion kernels ----------------

__global__ void cvt_f32_bf16(const float* __restrict__ src, __bf16* __restrict__ dst, int n) {
  int i = blockIdx.x * blockDim.x + threadIdx.x;
  if (i < n) dst[i] = (__bf16)src[i];
}

// src is [K x N] row-major f32, dst is [N x K] row-major bf16 (dst[n][k] = src[k][n])
__global__ void transpose_cvt(const float* __restrict__ src, __bf16* __restrict__ dst, int K, int N) {
  int i = blockIdx.x * blockDim.x + threadIdx.x;
  if (i < K * N) {
    int k = i / N, n = i % N;
    dst[(size_t)n * K + k] = (__bf16)src[i];
  }
}

// ---------------- LDS-tiled bf16 WMMA GEMM ----------------
// C[Mr x Nc] = A[Mr x K] @ B, with B supplied TRANSPOSED as Bt[Nc x K] (row n = column n of B).
// Workgroup (8 waves / 256 threads) computes a 128x128 macro-tile; per 64-K-step, A(128x64)
// and Bt(128x64) are staged in LDS (stride 72 keeps ds_load_b128 16B-aligned & spreads banks),
// each wave computes 32(M) x 64(N) = 8 16x16 accumulators -> 8x data reuse out of LDS.
// mode 0: f32 row-major out (+bias[col]); mode 1: bf16 row-major out (val*scale);
// mode 2: bf16 out stored transposed per batch: out[(b*Nc + col)*Mbatch + m], b = row/Mbatch.
__global__ void __launch_bounds__(256) gemm_bf16_wmma(
    const __bf16* __restrict__ A, const __bf16* __restrict__ Bt,
    void* __restrict__ Out, const float* __restrict__ bias,
    int Mr, int Nc, int K, float scale, int mode, int Mbatch) {
  __shared__ __align__(16) __bf16 ldsA[128][72];
  __shared__ __align__(16) __bf16 ldsB[128][72];
  const int tid  = threadIdx.x;
  const int lane = tid & 31;
  const int wave = tid >> 5;
  const int li = lane & 15, hi = lane >> 4;
  const int wm = wave & 3;                      // 4 M-slices of 32 rows
  const int wn = wave >> 2;                     // 2 N-slices of 64 cols
  const int nb = Nc >> 7;
  const int bm = blockIdx.x / nb, bn = blockIdx.x % nb;
  const int rbase = bm << 7, nbase = bn << 7;

  v8f acc[2][4];
  #pragma unroll
  for (int mi = 0; mi < 2; ++mi)
    #pragma unroll
    for (int ni = 0; ni < 4; ++ni) acc[mi][ni] = (v8f){};

  for (int k0 = 0; k0 < K; k0 += 64) {
    // cooperative stage: 128 rows x 64 bf16 = 1024 16B-chunks per matrix, 4 per thread
    #pragma unroll
    for (int i = 0; i < 4; ++i) {
      const int cid = tid + i * 256;
      const int row = cid >> 3, cc = (cid & 7) * 8;
      *(v8bf*)&ldsA[row][cc] = *(const v8bf*)(A  + (size_t)(rbase + row) * K + k0 + cc);
      *(v8bf*)&ldsB[row][cc] = *(const v8bf*)(Bt + (size_t)(nbase + row) * K + k0 + cc);
    }
    __syncthreads();
    #pragma unroll
    for (int ks = 0; ks < 64; ks += 32) {
      AFrag a0, a1;
      a0.h[0] = *(const v8bf*)&ldsA[wm * 32 +      li][ks +     hi * 16];
      a0.h[1] = *(const v8bf*)&ldsA[wm * 32 +      li][ks + 8 + hi * 16];
      a1.h[0] = *(const v8bf*)&ldsA[wm * 32 + 16 + li][ks +     hi * 16];
      a1.h[1] = *(const v8bf*)&ldsA[wm * 32 + 16 + li][ks + 8 + hi * 16];
      #pragma unroll
      for (int ni = 0; ni < 4; ++ni) {
        v16bf b = *(const v16bf*)&ldsB[wn * 64 + ni * 16 + li][ks + hi * 16];
        acc[0][ni] = __builtin_amdgcn_wmma_f32_16x16x32_bf16(false, a0.v, false, b, (short)0, acc[0][ni], false, false);
        acc[1][ni] = __builtin_amdgcn_wmma_f32_16x16x32_bf16(false, a1.v, false, b, (short)0, acc[1][ni], false, false);
      }
    }
    __syncthreads();
  }

  #pragma unroll
  for (int mi = 0; mi < 2; ++mi) {
    #pragma unroll
    for (int ni = 0; ni < 4; ++ni) {
      #pragma unroll
      for (int v = 0; v < 8; ++v) {
        const int row = rbase + wm * 32 + mi * 16 + v + 8 * hi;  // C/D: VGPR v -> M = v + 8*hi
        const int col = nbase + wn * 64 + ni * 16 + li;          //      lane li -> N = li
        float val = acc[mi][ni][v] * scale;
        if (mode == 0) {
          ((float*)Out)[(size_t)row * Nc + col] = val + (bias ? bias[col] : 0.0f);
        } else if (mode == 1) {
          ((__bf16*)Out)[(size_t)row * Nc + col] = (__bf16)val;
        } else {
          const int bb = row / Mbatch, mm = row % Mbatch;
          ((__bf16*)Out)[((size_t)bb * Nc + col) * Mbatch + mm] = (__bf16)val;
        }
      }
    }
  }
}

// ---------------- flash attention with WMMA ----------------
// One wave per (batch, head, 16-query tile). Loops over M=4096 keys in 32-key blocks.
// Qb: bf16 [B*N, INNER] with SCALE*log2e pre-folded (softmax done in base-2 domain);
// Kb: bf16 [B*M, INNER]; Vt: bf16 [B, INNER, M]; imp: f32 [B, M]; Ob: bf16 [B*N, INNER].
__global__ void __launch_bounds__(128) flash_attn_wmma(
    const __bf16* __restrict__ Qb, const __bf16* __restrict__ Kb,
    const __bf16* __restrict__ Vt, const float* __restrict__ imp,
    __bf16* __restrict__ Ob) {
  __shared__ __align__(16) __bf16 ldsP[4][16][40];   // per-wave 16x32 P tile, stride 40
  const int lane = threadIdx.x & 31;
  const int wave = threadIdx.x >> 5;
  const int li = lane & 15, hi = lane >> 4;
  const int tile = blockIdx.x * 4 + wave;            // B*HEADS*(NQ/16) = 2048 tiles
  if (tile >= BATCH * HEADS_ * (NQ / 16)) return;
  const int qt = tile & 63;
  const int h  = (tile >> 6) & (HEADS_ - 1);
  const int b  = tile >> 10;

  const __bf16* qrow = Qb + (size_t)(b * NQ + qt * 16 + li) * INNER_ + h * DHEAD;
  AFrag qa0, qa1;
  qa0.h[0] = *(const v8bf*)(qrow +      hi * 16);
  qa0.h[1] = *(const v8bf*)(qrow +  8 + hi * 16);
  qa1.h[0] = *(const v8bf*)(qrow + 32 + hi * 16);
  qa1.h[1] = *(const v8bf*)(qrow + 40 + hi * 16);

  const __bf16* kbase = Kb + (size_t)b * MKV * INNER_ + h * DHEAD;
  const __bf16* vbase = Vt + ((size_t)b * INNER_ + h * DHEAD) * MKV;
  const float*  impb  = imp + b * MKV;

  float mrun[8], lrun[8];
  v8f o0 = {}, o1 = {}, o2 = {}, o3 = {};
  #pragma unroll
  for (int v = 0; v < 8; ++v) { mrun[v] = -1e30f; lrun[v] = 0.0f; }

  for (int kb = 0; kb < MKV; kb += 32) {
    const __bf16* k0 = kbase + (size_t)(kb +      li) * INNER_;
    const __bf16* k1 = kbase + (size_t)(kb + 16 + li) * INNER_;
    if (kb + 32 < MKV) {                            // stream hint -> global_prefetch_b8
      __builtin_prefetch(kbase + (size_t)(kb + 32 + li) * INNER_, 0, 3);
      __builtin_prefetch(kbase + (size_t)(kb + 48 + li) * INNER_, 0, 3);
    }
    v8f s0 = {}, s1 = {};
    {
      v16bf bk = *(const v16bf*)(k0 +      hi * 16);
      s0 = __builtin_amdgcn_wmma_f32_16x16x32_bf16(false, qa0.v, false, bk, (short)0, s0, false, false);
      bk = *(const v16bf*)(k0 + 32 + hi * 16);
      s0 = __builtin_amdgcn_wmma_f32_16x16x32_bf16(false, qa1.v, false, bk, (short)0, s0, false, false);
      bk = *(const v16bf*)(k1 +      hi * 16);
      s1 = __builtin_amdgcn_wmma_f32_16x16x32_bf16(false, qa0.v, false, bk, (short)0, s1, false, false);
      bk = *(const v16bf*)(k1 + 32 + hi * 16);
      s1 = __builtin_amdgcn_wmma_f32_16x16x32_bf16(false, qa1.v, false, bk, (short)0, s1, false, false);
    }
    const float bias0 = impb[kb + li]      * LOG2E; // logits already in base-2 domain
    const float bias1 = impb[kb + 16 + li] * LOG2E;

    float p0[8], p1[8], alpha[8];
    #pragma unroll
    for (int v = 0; v < 8; ++v) {
      float x0 = s0[v] + bias0, x1 = s1[v] + bias1;
      float mx = fmaxf(x0, x1);
      mx = fmaxf(mx, __shfl_xor(mx, 1, 32));
      mx = fmaxf(mx, __shfl_xor(mx, 2, 32));
      mx = fmaxf(mx, __shfl_xor(mx, 4, 32));
      mx = fmaxf(mx, __shfl_xor(mx, 8, 32));
      float mnew = fmaxf(mrun[v], mx);
      float al = exp2f(mrun[v] - mnew);
      float e0 = exp2f(x0 - mnew);
      float e1 = exp2f(x1 - mnew);
      float rs = e0 + e1;
      rs += __shfl_xor(rs, 1, 32);
      rs += __shfl_xor(rs, 2, 32);
      rs += __shfl_xor(rs, 4, 32);
      rs += __shfl_xor(rs, 8, 32);
      lrun[v] = lrun[v] * al + rs;
      mrun[v] = mnew;
      alpha[v] = al; p0[v] = e0; p1[v] = e1;
    }
    #pragma unroll
    for (int v = 0; v < 8; ++v) {
      o0[v] *= alpha[v]; o1[v] *= alpha[v]; o2[v] *= alpha[v]; o3[v] *= alpha[v];
    }

    // P (C/D layout) -> LDS -> A-operand layout
    __bf16 (*pb)[40] = ldsP[wave];
    #pragma unroll
    for (int v = 0; v < 8; ++v) {
      pb[v + 8 * hi][li]      = (__bf16)p0[v];
      pb[v + 8 * hi][16 + li] = (__bf16)p1[v];
    }
    asm volatile("s_wait_dscnt 0" ::: "memory");     // LDS store -> load hazard within wave
    AFrag pa;
    pa.h[0] = *(const v8bf*)(&pb[li][hi * 16]);
    pa.h[1] = *(const v8bf*)(&pb[li][8 + hi * 16]);

    v16bf bv;
    bv = *(const v16bf*)(vbase + (size_t)( 0 + li) * MKV + kb + hi * 16);
    o0 = __builtin_amdgcn_wmma_f32_16x16x32_bf16(false, pa.v, false, bv, (short)0, o0, false, false);
    bv = *(const v16bf*)(vbase + (size_t)(16 + li) * MKV + kb + hi * 16);
    o1 = __builtin_amdgcn_wmma_f32_16x16x32_bf16(false, pa.v, false, bv, (short)0, o1, false, false);
    bv = *(const v16bf*)(vbase + (size_t)(32 + li) * MKV + kb + hi * 16);
    o2 = __builtin_amdgcn_wmma_f32_16x16x32_bf16(false, pa.v, false, bv, (short)0, o2, false, false);
    bv = *(const v16bf*)(vbase + (size_t)(48 + li) * MKV + kb + hi * 16);
    o3 = __builtin_amdgcn_wmma_f32_16x16x32_bf16(false, pa.v, false, bv, (short)0, o3, false, false);
  }

  #pragma unroll
  for (int v = 0; v < 8; ++v) {
    const size_t rbase = (size_t)(b * NQ + qt * 16 + v + 8 * hi) * INNER_ + h * DHEAD + li;
    const float inv = 1.0f / lrun[v];
    Ob[rbase +  0] = (__bf16)(o0[v] * inv);
    Ob[rbase + 16] = (__bf16)(o1[v] * inv);
    Ob[rbase + 32] = (__bf16)(o2[v] * inv);
    Ob[rbase + 48] = (__bf16)(o3[v] * inv);
  }
}

// ---------------- host orchestration ----------------

extern "C" void kernel_launch(void* const* d_in, const int* in_sizes, int n_in,
                              void* d_out, int out_size, void* d_ws, size_t ws_size,
                              hipStream_t stream) {
  (void)in_sizes; (void)n_in; (void)out_size; (void)ws_size;
  const float* x   = (const float*)d_in[0];
  const float* ctx = (const float*)d_in[1];
  const float* imp = (const float*)d_in[2];
  const float* Wq  = (const float*)d_in[3];
  const float* Wk  = (const float*)d_in[4];
  const float* Wv  = (const float*)d_in[5];
  const float* Wo  = (const float*)d_in[6];
  const float* bo  = (const float*)d_in[7];
  float* out = (float*)d_out;

  char* ws = (char*)d_ws;
  const size_t MB = 1024 * 1024;
  __bf16* xb  = (__bf16*)(ws +  0 * MB);   //  4 MB  x bf16
  __bf16* cb  = (__bf16*)(ws +  4 * MB);   // 16 MB  context bf16
  __bf16* wqt = (__bf16*)(ws + 20 * MB);   //  2 MB  Wq^T bf16
  __bf16* wkt = (__bf16*)(ws + 22 * MB);   //  2 MB  Wk^T bf16
  __bf16* wvt = (__bf16*)(ws + 24 * MB);   //  2 MB  Wv^T bf16
  __bf16* wot = (__bf16*)(ws + 26 * MB);   //  2 MB  Wo^T bf16
  __bf16* Qb  = (__bf16*)(ws + 28 * MB);   //  4 MB  Q bf16 (SCALE*log2e folded)
  __bf16* Kb  = (__bf16*)(ws + 32 * MB);   // 16 MB  K bf16
  __bf16* Vt  = (__bf16*)(ws + 48 * MB);   // 16 MB  V bf16, [B][INNER][M]
  __bf16* Ab  = (__bf16*)(ws + 64 * MB);   //  4 MB  attention output bf16

  const int nx = BATCH * NQ  * QDIM;
  const int nc = BATCH * MKV * CDIM;
  const int nw = QDIM * INNER_;
  cvt_f32_bf16<<<(nx + 255) / 256, 256, 0, stream>>>(x,   xb, nx);
  cvt_f32_bf16<<<(nc + 255) / 256, 256, 0, stream>>>(ctx, cb, nc);
  transpose_cvt<<<(nw + 255) / 256, 256, 0, stream>>>(Wq, wqt, QDIM,   INNER_);
  transpose_cvt<<<(nw + 255) / 256, 256, 0, stream>>>(Wk, wkt, CDIM,   INNER_);
  transpose_cvt<<<(nw + 255) / 256, 256, 0, stream>>>(Wv, wvt, CDIM,   INNER_);
  transpose_cvt<<<(nw + 255) / 256, 256, 0, stream>>>(Wo, wot, INNER_, QDIM);

  // Q = (x @ Wq) * SCALE * log2e  -> bf16 [2048, 1024]
  {
    const int Mr = BATCH * NQ, Nc = INNER_, K = QDIM;
    gemm_bf16_wmma<<<(Mr / 128) * (Nc / 128), 256, 0, stream>>>(
        xb, wqt, Qb, nullptr, Mr, Nc, K, SCALE_ * LOG2E, 1, 1);
  }
  // K = context @ Wk  -> bf16 [8192, 1024]
  {
    const int Mr = BATCH * MKV, Nc = INNER_, K = CDIM;
    gemm_bf16_wmma<<<(Mr / 128) * (Nc / 128), 256, 0, stream>>>(
        cb, wkt, Kb, nullptr, Mr, Nc, K, 1.0f, 1, 1);
  }
  // V = context @ Wv  -> bf16 transposed per batch [B][1024][4096]
  {
    const int Mr = BATCH * MKV, Nc = INNER_, K = CDIM;
    gemm_bf16_wmma<<<(Mr / 128) * (Nc / 128), 256, 0, stream>>>(
        cb, wvt, Vt, nullptr, Mr, Nc, K, 1.0f, 2, MKV);
  }
  // flash attention: 2048 (b, h, qtile) waves, 4 waves/block
  flash_attn_wmma<<<512, 128, 0, stream>>>(Qb, Kb, Vt, imp, Ab);
  // out = attn_out @ Wo + bo  -> f32 [2048, 1024]
  {
    const int Mr = BATCH * NQ, Nc = QDIM, K = INNER_;
    gemm_bf16_wmma<<<(Mr / 128) * (Nc / 128), 256, 0, stream>>>(
        Ab, wot, out, bo, Mr, Nc, K, 1.0f, 0, 1);
  }
}
